// LSTM_20710332301579
// MI455X (gfx1250) — matrix operand order repaired and checked by reading, not compiled, not run on hardware
//
#include <hip/hip_runtime.h>
#include <hip/hip_bf16.h>

typedef __attribute__((ext_vector_type(16))) _Float16 v16h;
typedef __attribute__((ext_vector_type(8)))  _Float16 v8h;
typedef __attribute__((ext_vector_type(8)))  float    v8f;

#define HID      128
#define G4H      512          // 4*HID
#define T_STEPS  512
#define BATCH    256
#define BTILE    16           // batch rows per workgroup
#define NWG      (BATCH / BTILE)
#define THREADS  256          // 8 waves (wave32)
#define NT_PER_W 4            // N-tiles (16 cols) per wave: 8*4*16 = 512 gate cols
#define KK_HH    (HID / 32)   // 4 recurrent k-steps
#define LDH      (HID + 8)    // padded LDS row stride (halfs) for K=128
#define LDG      (G4H + 4)    // padded gate-buffer stride (floats)

__device__ __forceinline__ float sigmoidf_(float x) {
    return 1.0f / (1.0f + __expf(-x));
}
__device__ __forceinline__ float tanhf_(float x) {
    return 1.0f - 2.0f / (1.0f + __expf(2.0f * x));
}

// A fragment: 16x32 f16, M = lane%16.
// lanes 0-15 hold K = 0..15 (contiguous); lanes 16-31 hold K = 16..31.
__device__ __forceinline__ v16h load_A_frag(const _Float16* sA, int lane, int kk, int ldk) {
    int m     = lane & 15;
    int kbase = kk * 32 + ((lane >> 4) << 4);
    const _Float16* p = sA + m * ldk + kbase;
    v8h lo = *(const v8h*)(p);        // K_local kbase .. kbase+7
    v8h hi = *(const v8h*)(p + 8);    // K_local kbase+8 .. kbase+15
    return __builtin_shufflevector(lo, hi, 0, 1, 2, 3, 4, 5, 6, 7,
                                           8, 9, 10, 11, 12, 13, 14, 15);
}

// B fragment from LDS: 32x16 f16, N = n0 + lane%16.
// lanes 0-15 hold K 0..15 (contiguous), lanes 16-31 hold K 16..31.
__device__ __forceinline__ v16h load_B_frag(const _Float16* sW, int n0, int lane, int kk, int ldk) {
    int n    = n0 + (lane & 15);
    int koff = kk * 32 + ((lane >> 4) << 4);
    return *(const v16h*)(sW + n * ldk + koff);
}

// B fragment loaded directly from global f32 [4H, H] row-major, converted to f16.
// Loop-invariant: done once, lives in VGPRs for the whole time loop.
__device__ __forceinline__ v16h load_B_frag_global_f32(const float* __restrict__ W,
                                                       int n0, int lane, int kk, int ldk) {
    int n    = n0 + (lane & 15);
    int koff = kk * 32 + ((lane >> 4) << 4);
    const float* wp = W + (size_t)n * ldk + koff;
    v16h frag;
#pragma unroll
    for (int i = 0; i < 16; ++i) frag[i] = (_Float16)wp[i];
    return frag;
}

// One LSTM layer, persistent time loop. Each workgroup owns BTILE batch rows.
// x: [BATCH, T, K_IN] (float for layer 0, f16 for layer 1)
// W_ih: [4H, K_IN] f32, W_hh: [4H, H] f32, biases [4H] f32.
// hseq (f16 [BATCH,T,H]) written if non-null; hlast (f32 [BATCH,H]) written at t=T-1 if non-null.
template <int K_IN, typename XT>
__global__ __launch_bounds__(THREADS)
void lstm_layer_kernel(const XT* __restrict__ x,
                       const float* __restrict__ W_ih,
                       const float* __restrict__ W_hh,
                       const float* __restrict__ b_ih,
                       const float* __restrict__ b_hh,
                       _Float16* __restrict__ hseq,
                       float* __restrict__ hlast) {
    constexpr int LDKI  = K_IN + 8;
    constexpr int KK_IN = K_IN / 32;

    extern __shared__ char smem[];
    _Float16* sWih = (_Float16*)smem;                 // [G4H][LDKI]
    _Float16* sX0  = sWih + G4H * LDKI;               // [BTILE][LDKI] (double buffered)
    _Float16* sX1  = sX0 + BTILE * LDKI;              // [BTILE][LDKI]
    _Float16* sH   = sX1 + BTILE * LDKI;              // [BTILE][LDH]
    float*    sG   = (float*)(sH + BTILE * LDH);      // [BTILE][LDG]

    const int tid  = threadIdx.x;
    const int lane = tid & 31;
    const int wave = tid >> 5;
    const int b0   = blockIdx.x * BTILE;

    // ---- Stage W_ih to LDS as f16 (once) ----
    for (int idx = tid; idx < G4H * K_IN; idx += THREADS) {
        int n = idx / K_IN, k = idx - n * K_IN;
        sWih[n * LDKI + k] = (_Float16)W_ih[idx];
    }
    // h0 = 0
    for (int idx = tid; idx < BTILE * LDH; idx += THREADS) sH[idx] = (_Float16)0.0f;
    // Stage x_0 into buffer 0
    for (int idx = tid; idx < BTILE * K_IN; idx += THREADS) {
        int r = idx / K_IN, k = idx - r * K_IN;
        sX0[r * LDKI + k] = (_Float16)x[((size_t)(b0 + r) * T_STEPS) * K_IN + k];
    }

    // ---- Loop-invariant W_hh B-fragments live in registers (128 VGPRs/wave) ----
    v16h whh[KK_HH][NT_PER_W];
#pragma unroll
    for (int kk = 0; kk < KK_HH; ++kk)
#pragma unroll
        for (int nt = 0; nt < NT_PER_W; ++nt)
            whh[kk][nt] = load_B_frag_global_f32(
                W_hh, (wave * NT_PER_W + nt) * 16, lane, kk, HID);

    // ---- Bias sums for the elementwise phase (per-thread columns) ----
    const int em = tid >> 4;   // batch row handled in elementwise phase
    const int ej = tid & 15;
    float bI[8], bF[8], bG[8], bO[8];
#pragma unroll
    for (int q = 0; q < 8; ++q) {
        int j = ej + 16 * q;
        bI[q] = b_ih[j]           + b_hh[j];
        bF[q] = b_ih[HID + j]     + b_hh[HID + j];
        bG[q] = b_ih[2 * HID + j] + b_hh[2 * HID + j];
        bO[q] = b_ih[3 * HID + j] + b_hh[3 * HID + j];
    }

    float creg[8];
#pragma unroll
    for (int q = 0; q < 8; ++q) creg[q] = 0.0f;

    __syncthreads();

    for (int t = 0; t < T_STEPS; ++t) {
        const _Float16* sXcur = (t & 1) ? sX1 : sX0;
        _Float16*       sXnxt = (t & 1) ? sX0 : sX1;

        // ---- Gate pre-activations: g = x_t @ W_ih^T + h @ W_hh^T (bias later) ----
        v8f acc[NT_PER_W];
#pragma unroll
        for (int nt = 0; nt < NT_PER_W; ++nt)
            acc[nt] = (v8f){0.f, 0.f, 0.f, 0.f, 0.f, 0.f, 0.f, 0.f};
#pragma unroll
        for (int kk = 0; kk < KK_IN; ++kk) {
            v16h a = load_A_frag(sXcur, lane, kk, LDKI);
#pragma unroll
            for (int nt = 0; nt < NT_PER_W; ++nt) {
                int n0 = (wave * NT_PER_W + nt) * 16;
                v16h b = load_B_frag(sWih, n0, lane, kk, LDKI);
                acc[nt] = __builtin_amdgcn_wmma_f32_16x16x32_f16(
                    false, a, false, b, (short)0, acc[nt], false, false);
            }
        }
#pragma unroll
        for (int kk = 0; kk < KK_HH; ++kk) {
            v16h a = load_A_frag(sH, lane, kk, LDH);
#pragma unroll
            for (int nt = 0; nt < NT_PER_W; ++nt) {
                acc[nt] = __builtin_amdgcn_wmma_f32_16x16x32_f16(
                    false, a, false, whh[kk][nt], (short)0, acc[nt], false, false);
            }
        }
        // Scatter C tiles: M = v + 8*(lane>>4), N = n0 + lane%16
#pragma unroll
        for (int nt = 0; nt < NT_PER_W; ++nt) {
            int nn    = (wave * NT_PER_W + nt) * 16 + (lane & 15);
            int mbase = (lane >> 4) << 3;
#pragma unroll
            for (int v = 0; v < 8; ++v) {
                sG[(mbase + v) * LDG + nn] = acc[nt][v];
            }
        }
        __syncthreads();   // sG ready

        // ---- Elementwise gates + state update; also stage x_{t+1} ----
        const float* gr = sG + em * LDG;
#pragma unroll
        for (int q = 0; q < 8; ++q) {
            int j = ej + 16 * q;
            float ig = sigmoidf_(gr[j] + bI[q]);
            float fg = sigmoidf_(gr[128 + j] + bF[q]);
            float gg = tanhf_(gr[256 + j] + bG[q]);
            float og = sigmoidf_(gr[384 + j] + bO[q]);
            creg[q] = fg * creg[q] + ig * gg;
            float hv = og * tanhf_(creg[q]);
            sH[em * LDH + j] = (_Float16)hv;
            if (hseq)
                hseq[((size_t)(b0 + em) * T_STEPS + t) * HID + j] = (_Float16)hv;
            if (hlast && t == T_STEPS - 1)
                hlast[(size_t)(b0 + em) * HID + j] = hv;
        }
        if (t + 1 < T_STEPS) {
            for (int idx = tid; idx < BTILE * K_IN; idx += THREADS) {
                int r = idx / K_IN, k = idx - r * K_IN;
                sXnxt[r * LDKI + k] =
                    (_Float16)x[((size_t)(b0 + r) * T_STEPS + (t + 1)) * K_IN + k];
            }
            if (t + 2 < T_STEPS) {
                const XT* nx = x + ((size_t)(b0 + (tid & 15)) * T_STEPS + (t + 2)) * K_IN;
                __builtin_prefetch(nx, 0, 0);   // global_prefetch_b8 two steps ahead
            }
        }
        __syncthreads();   // sH + sXnxt ready for next step; sG safe to overwrite
    }
}

// Tiny FC head: out = relu(relu(h_last @ fc1_w^T + fc1_b) @ fc_w^T + fc_b), [256,1]
__global__ __launch_bounds__(256)
void fc_head_kernel(const float* __restrict__ hlast,
                    const float* __restrict__ fc1_w, const float* __restrict__ fc1_b,
                    const float* __restrict__ fc_w, const float* __restrict__ fc_b,
                    float* __restrict__ out) {
    int b = blockIdx.x * blockDim.x + threadIdx.x;
    if (b >= BATCH) return;
    const float* hr = hlast + (size_t)b * HID;
    float acc2 = 0.0f;
#pragma unroll 4
    for (int u = 0; u < HID / 2; ++u) {
        float a = fc1_b[u];
        const float* wr = fc1_w + u * HID;
#pragma unroll 8
        for (int k = 0; k < HID; ++k) a += hr[k] * wr[k];
        a = fmaxf(a, 0.0f);
        acc2 += a * fc_w[u];
    }
    out[b] = fmaxf(acc2 + fc_b[0], 0.0f);
}

static size_t lstm_smem_bytes(int k_in) {
    size_t ldki  = (size_t)k_in + 8;
    size_t halfs = (size_t)G4H * ldki        // sWih
                 + 2 * (size_t)BTILE * ldki  // sX double buffer
                 + (size_t)BTILE * LDH;      // sH
    return halfs * 2 + (size_t)BTILE * LDG * 4;   // + sG (f32)
}

extern "C" void kernel_launch(void* const* d_in, const int* in_sizes, int n_in,
                              void* d_out, int out_size, void* d_ws, size_t ws_size,
                              hipStream_t stream) {
    const float* x     = (const float*)d_in[0];
    const float* W_ih0 = (const float*)d_in[1];
    const float* W_hh0 = (const float*)d_in[2];
    const float* b_ih0 = (const float*)d_in[3];
    const float* b_hh0 = (const float*)d_in[4];
    const float* W_ih1 = (const float*)d_in[5];
    const float* W_hh1 = (const float*)d_in[6];
    const float* b_ih1 = (const float*)d_in[7];
    const float* b_hh1 = (const float*)d_in[8];
    const float* fc1_w = (const float*)d_in[9];
    const float* fc1_b = (const float*)d_in[10];
    const float* fc_w  = (const float*)d_in[11];
    const float* fc_b  = (const float*)d_in[12];
    float* out = (float*)d_out;

    // Workspace: f16 hidden sequence of layer 0, then f32 last-h of layer 1.
    _Float16* h1seq = (_Float16*)d_ws;                                   // 256*512*128 halfs = 32 MB
    float*    hlast = (float*)((char*)d_ws + (size_t)BATCH * T_STEPS * HID * 2);

    size_t smem0 = lstm_smem_bytes(64);    // ~116 KB
    size_t smem1 = lstm_smem_bytes(128);   // ~185 KB
    (void)hipFuncSetAttribute((const void*)lstm_layer_kernel<64, float>,
                              hipFuncAttributeMaxDynamicSharedMemorySize, (int)smem0);
    (void)hipFuncSetAttribute((const void*)lstm_layer_kernel<128, _Float16>,
                              hipFuncAttributeMaxDynamicSharedMemorySize, (int)smem1);

    // Layer 0: x (f32) -> h1seq (f16). No last-h needed.
    lstm_layer_kernel<64, float><<<NWG, THREADS, smem0, stream>>>(
        x, W_ih0, W_hh0, b_ih0, b_hh0, h1seq, nullptr);

    // Layer 1: h1seq (f16) -> hlast (f32). No sequence output needed.
    lstm_layer_kernel<128, _Float16><<<NWG, THREADS, smem1, stream>>>(
        h1seq, W_ih1, W_hh1, b_ih1, b_hh1, nullptr, hlast);

    // FC head.
    fc_head_kernel<<<1, 256, 0, stream>>>(hlast, fc1_w, fc1_b, fc_w, fc_b, out);
    (void)in_sizes; (void)n_in; (void)out_size; (void)ws_size;
}
